// GCN_age_64544768525182
// MI455X (gfx1250) — matrix operand order repaired and checked by reading, not compiled, not run on hardware
//
#include <hip/hip_runtime.h>

typedef __attribute__((ext_vector_type(16))) __bf16 v16bf;
typedef __attribute__((ext_vector_type(8)))  float  v8f;

// ---------------- utility: zero a float buffer (grid-stride) ----------------
__global__ void k_zero(float* __restrict__ p, long long n) {
  long long i = (long long)blockIdx.x * blockDim.x + threadIdx.x;
  long long stride = (long long)gridDim.x * blockDim.x;
  for (; i < n; i += stride) p[i] = 0.0f;
}

// ---------------- degree / dinv ----------------
__global__ void k_deg(const int* __restrict__ dst, float* __restrict__ deg, int E) {
  int e = blockIdx.x * blockDim.x + threadIdx.x;
  if (e < E) atomicAdd(&deg[dst[e]], 1.0f);
}
__global__ void k_dinv(float* __restrict__ deg, int N) {
  int n = blockIdx.x * blockDim.x + threadIdx.x;
  if (n < N) deg[n] = rsqrtf(deg[n] + 1.0f);   // self-loop: deg+1
}

// ---------------- pack A (fp32 row-major [M,K]) -> WMMA A-fragment bf16 -----
// Layout: Ap[((mtile*KBLKS)+kblk)*512 + lane*16 + e]
// A frag (16x32): row = mtile*16 + (lane&15), half = lane>>4,
//                 k = kblk*32 + e + 16*half
__global__ void k_pack_a(const float* __restrict__ A, __bf16* __restrict__ Ap,
                         int K, int KBLKS) {
  const int mtile = blockIdx.x, kblk = blockIdx.y;
  const int t = threadIdx.x;              // 0..511
  const int lane = t >> 4, e = t & 15;
  const int half = lane >> 4, lm = lane & 15;
  const int row = mtile * 16 + lm;
  const int k = kblk * 32 + e + 16 * half;
  const float v = (k < K) ? A[(size_t)row * K + k] : 0.0f;
  Ap[((size_t)mtile * KBLKS + kblk) * 512 + t] = (__bf16)v;
}

// ---------------- pack W (fp32 row-major [K,N]) -> WMMA B-fragment bf16 -----
// Layout: Bp[((ntile*KBLKS)+kblk)*512 + lane*16 + e]
// B frag (32x16): col = ntile*16 + (lane&15), k = kblk*32 + e + 16*(lane>>4)
__global__ void k_pack_b(const float* __restrict__ W, __bf16* __restrict__ Bp,
                         int K, int N, int KBLKS) {
  const int ntile = blockIdx.x, kblk = blockIdx.y;
  const int t = threadIdx.x;              // 0..511
  const int lane = t >> 4, e = t & 15;
  const int col = ntile * 16 + (lane & 15);
  const int k = kblk * 32 + e + 16 * (lane >> 4);
  const float v = ((k < K) && (col < N)) ? W[(size_t)k * N + col] : 0.0f;
  Bp[((size_t)ntile * KBLKS + kblk) * 512 + t] = (__bf16)v;
}

// ---------------- tile store helper (epilogue only) ----------------
__device__ __forceinline__ void store_tile(float* __restrict__ C, const v8f& acc,
                                           const float* __restrict__ bias,
                                           int mtile, int ntile, int N,
                                           int half, int lm, int relu) {
  const int col = ntile * 16 + lm;
  if (col < N) {
    const float bv = bias ? bias[col] : 0.0f;
#pragma unroll
    for (int r = 0; r < 8; ++r) {
      float v = acc[r] + bv;
      if (relu) v = fmaxf(v, 0.0f);
      C[(size_t)(mtile * 16 + r + 8 * half) * N + col] = v;
    }
  }
}

// ---------------- WMMA GEMM on packed fragments, 2x2 tiles per wave ---------
// 4 v_wmma per 8 b128 loads: halves L2 fragment traffic vs 1 tile/wave.
__global__ void k_gemm_wmma(const __bf16* __restrict__ Ap, const __bf16* __restrict__ Bp,
                            float* __restrict__ C, const float* __restrict__ bias,
                            int N, int KBLKS, int MT, int NT, int relu) {
  const int lane = threadIdx.x & 31;
  const int half = lane >> 4;
  const int lm   = lane & 15;
  const int m0 = blockIdx.x * 2;
  const int n0 = blockIdx.y * 2;
  const int m1 = (m0 + 1 < MT) ? m0 + 1 : m0;   // clamp: loads stay in-bounds
  const int n1 = (n0 + 1 < NT) ? n0 + 1 : n0;

  const __bf16* ap0 = Ap + ((size_t)m0 * KBLKS) * 512 + lane * 16;
  const __bf16* ap1 = Ap + ((size_t)m1 * KBLKS) * 512 + lane * 16;
  const __bf16* bp0 = Bp + ((size_t)n0 * KBLKS) * 512 + lane * 16;
  const __bf16* bp1 = Bp + ((size_t)n1 * KBLKS) * 512 + lane * 16;

  v8f acc00 = {0.f,0.f,0.f,0.f,0.f,0.f,0.f,0.f};
  v8f acc01 = acc00, acc10 = acc00, acc11 = acc00;

  for (int kb = 0; kb < KBLKS; ++kb) {
    v16bf a0 = *reinterpret_cast<const v16bf*>(ap0);
    v16bf a1 = *reinterpret_cast<const v16bf*>(ap1);
    v16bf b0 = *reinterpret_cast<const v16bf*>(bp0);
    v16bf b1 = *reinterpret_cast<const v16bf*>(bp1);
    __builtin_prefetch(ap0 + 512, 0, 1);     // global_prefetch next k-block
    __builtin_prefetch(bp0 + 512, 0, 1);
    acc00 = __builtin_amdgcn_wmma_f32_16x16x32_bf16(false, a0, false, b0,
                                                    (short)0, acc00, false, false);
    acc01 = __builtin_amdgcn_wmma_f32_16x16x32_bf16(false, a0, false, b1,
                                                    (short)0, acc01, false, false);
    acc10 = __builtin_amdgcn_wmma_f32_16x16x32_bf16(false, a1, false, b0,
                                                    (short)0, acc10, false, false);
    acc11 = __builtin_amdgcn_wmma_f32_16x16x32_bf16(false, a1, false, b1,
                                                    (short)0, acc11, false, false);
    ap0 += 512; ap1 += 512; bp0 += 512; bp1 += 512;
  }

  store_tile(C, acc00, bias, m0, n0, N, half, lm, relu);
  if (n1 != n0)              store_tile(C, acc01, bias, m0, n1, N, half, lm, relu);
  if (m1 != m0)              store_tile(C, acc10, bias, m1, n0, N, half, lm, relu);
  if (m1 != m0 && n1 != n0)  store_tile(C, acc11, bias, m1, n1, N, half, lm, relu);
}

// ---------------- edge scatter: H[dst] += coef * T[src] ----------------
__global__ void k_edge_scatter(const float* __restrict__ T, const int* __restrict__ src,
                               const int* __restrict__ dst, const float* __restrict__ dinv,
                               float* __restrict__ H, int Cch) {
  const int e = blockIdx.x;
  const int t = threadIdx.x;
  const int chunks = Cch >> 2;           // Cch divisible by 4 (500/400/300)
  if (t >= chunks) return;
  const int s = src[e], d = dst[e];
  const float coef = dinv[s] * dinv[d];
  const float4 v = *reinterpret_cast<const float4*>(T + (size_t)s * Cch + 4 * t);
  float* out = H + (size_t)d * Cch + 4 * t;
  atomicAdd(out + 0, v.x * coef);
  atomicAdd(out + 1, v.y * coef);
  atomicAdd(out + 2, v.z * coef);
  atomicAdd(out + 3, v.w * coef);
}

// ---------------- self-loop + bias + relu (in place on H) ----------------
__global__ void k_finalize(const float* __restrict__ T, const float* __restrict__ dinv,
                           const float* __restrict__ bias, float* __restrict__ H, int Cch) {
  const int n = blockIdx.x;
  const float dsq = dinv[n] * dinv[n];
  for (int c = threadIdx.x; c < Cch; c += blockDim.x) {
    size_t i = (size_t)n * Cch + c;
    float v = H[i] + T[i] * dsq + bias[c];
    H[i] = fmaxf(v, 0.0f);
  }
}

// ---------------- segment max over graphs (values >= 0 post-relu) -----------
__global__ void k_segmax(const float* __restrict__ H, const int* __restrict__ batch,
                         unsigned int* __restrict__ G, int Cch) {
  const int n = blockIdx.x;
  const int g = batch[n];
  for (int c = threadIdx.x; c < Cch; c += blockDim.x) {
    float v = H[(size_t)n * Cch + c];
    atomicMax(&G[(size_t)g * Cch + c], __float_as_uint(v));  // monotone for v>=0
  }
}

// ---------------- softmax over axis 0 (256 graphs x 6 classes) --------------
__global__ void k_softmax0(const float* __restrict__ logits, float* __restrict__ out) {
  __shared__ float red[256];
  const int t = threadIdx.x;             // 0..255 = graph id
  for (int c = 0; c < 6; ++c) {
    float v = logits[t * 6 + c];
    red[t] = v; __syncthreads();
    for (int s2 = 128; s2 > 0; s2 >>= 1) {
      if (t < s2) red[t] = fmaxf(red[t], red[t + s2]);
      __syncthreads();
    }
    float mx = red[0]; __syncthreads();
    float ev = __expf(v - mx);
    red[t] = ev; __syncthreads();
    for (int s2 = 128; s2 > 0; s2 >>= 1) {
      if (t < s2) red[t] += red[t + s2];
      __syncthreads();
    }
    float sum = red[0]; __syncthreads();
    out[t * 6 + c] = ev / sum;
  }
}

extern "C" void kernel_launch(void* const* d_in, const int* in_sizes, int n_in,
                              void* d_out, int out_size, void* d_ws, size_t ws_size,
                              hipStream_t stream) {
  const float* x    = (const float*)d_in[0];
  const int*   eidx = (const int*)d_in[1];   // [2, E] (JAX x64 off -> int32)
  const int*   batch= (const int*)d_in[2];
  const float* W1 = (const float*)d_in[3];   const float* b1 = (const float*)d_in[4];
  const float* W2 = (const float*)d_in[5];   const float* b2 = (const float*)d_in[6];
  const float* W3 = (const float*)d_in[7];   const float* b3 = (const float*)d_in[8];
  const float* Wl1 = (const float*)d_in[9];  const float* bl1 = (const float*)d_in[10];
  const float* Wl2 = (const float*)d_in[11]; const float* bl2 = (const float*)d_in[12];
  const float* Wl3 = (const float*)d_in[13]; const float* bl3 = (const float*)d_in[14];

  const int Nn = 50000, E = 800000, Gn = 256;
  const int* src = eidx;
  const int* dst = eidx + E;

  float* ws = (float*)d_ws;
  const size_t CAP = (size_t)Nn * 512;       // per rotating fp32 buffer (floats)
  float* bufA = ws;                           // GEMM outputs T
  float* bufB = ws + CAP;                     // H1, later H3
  float* bufC = ws + 2 * CAP;                 // H2
  float* dinv = ws + 3 * CAP;                 // [Nn]
  float* G    = dinv + Nn;                    // [256,300]
  float* F1   = G  + (size_t)Gn * 300;        // [256,200]
  float* F2   = F1 + (size_t)Gn * 200;        // [256,100]
  float* LG   = F2 + (size_t)Gn * 100;        // [256,6]
  // bf16 packed fragment buffers (aligned: all offsets are multiples of 16 floats)
  __bf16* Ap  = (__bf16*)(LG + 2048);         // cap: 3125*16*512 bf16 = 51.2 MB
  __bf16* Bp  = (__bf16*)((float*)Ap + (size_t)3125 * 16 * 256); // cap: 32*16*512 bf16

  auto zero = [&](float* p, long long n) {
    long long b = (n + 255) / 256;
    int blocks = (b > 32768) ? 32768 : (int)((b < 1) ? 1 : b);
    k_zero<<<blocks, 256, 0, stream>>>(p, n);
  };

  // packed-fragment GEMM: C = A[M,K] @ W[K,N] (+bias)(+relu)
  auto gemm = [&](const float* A, const float* W, float* C, const float* bias,
                  int M, int K, int N, int relu) {
    const int MT = M / 16;                 // M is always a multiple of 16
    const int KB = (K + 31) / 32;
    const int NT = (N + 15) / 16;
    k_pack_a<<<dim3(MT, KB), 512, 0, stream>>>(A, Ap, K, KB);
    k_pack_b<<<dim3(NT, KB), 512, 0, stream>>>(W, Bp, K, N, KB);
    k_gemm_wmma<<<dim3((MT + 1) / 2, (NT + 1) / 2), 32, 0, stream>>>(
        Ap, Bp, C, bias, N, KB, MT, NT, relu);
  };

  // --- degrees -> dinv ---
  zero(dinv, Nn);
  k_deg<<<(E + 255) / 256, 256, 0, stream>>>(dst, dinv, E);
  k_dinv<<<(Nn + 255) / 256, 256, 0, stream>>>(dinv, Nn);

  // --- layer 1: T = x@W1 (19->500); H1 = agg(T) + selfloop + b1, relu ---
  gemm(x, W1, bufA, nullptr, Nn, 19, 500, 0);
  zero(bufB, (long long)Nn * 500);
  k_edge_scatter<<<E, 128, 0, stream>>>(bufA, src, dst, dinv, bufB, 500);
  k_finalize<<<Nn, 256, 0, stream>>>(bufA, dinv, b1, bufB, 500);

  // --- layer 2: 500 -> 400 ---
  gemm(bufB, W2, bufA, nullptr, Nn, 500, 400, 0);
  zero(bufC, (long long)Nn * 400);
  k_edge_scatter<<<E, 128, 0, stream>>>(bufA, src, dst, dinv, bufC, 400);
  k_finalize<<<Nn, 256, 0, stream>>>(bufA, dinv, b2, bufC, 400);

  // --- layer 3: 400 -> 300 ---
  gemm(bufC, W3, bufA, nullptr, Nn, 400, 300, 0);
  zero(bufB, (long long)Nn * 300);
  k_edge_scatter<<<E, 128, 0, stream>>>(bufA, src, dst, dinv, bufB, 300);
  k_finalize<<<Nn, 256, 0, stream>>>(bufA, dinv, b3, bufB, 300);

  // --- per-graph max pooling ---
  zero(G, (long long)Gn * 300);
  k_segmax<<<Nn, 256, 0, stream>>>(bufB, batch, (unsigned int*)G, 300);

  // --- MLP head (WMMA with fused bias+relu) ---
  gemm(G,  Wl1, F1, bl1, Gn, 300, 200, 1);
  gemm(F1, Wl2, F2, bl2, Gn, 200, 100, 1);
  gemm(F2, Wl3, LG, bl3, Gn, 100, 6, 0);

  // --- softmax over graph axis ---
  k_softmax0<<<1, 256, 0, stream>>>(LG, (float*)d_out);
}
